// spatialAttentionGCN_88931592831704
// MI455X (gfx1250) — matrix operand order repaired and checked by reading, not compiled, not run on hardware
//
#include <hip/hip_runtime.h>
#include <hip/hip_bf16.h>

typedef __attribute__((ext_vector_type(16))) _Float16 v16h;
typedef __attribute__((ext_vector_type(8)))  _Float16 v8h;
typedef __attribute__((ext_vector_type(8)))  float    v8f;

#define N_NODES 2000
#define FD 64
#define TT 24
#define NTILES (N_NODES / 16)   // 125

union Frag16 { v16h v; v8h h[2]; };

// ---------------------------------------------------------------------------
// diag_w[i] = sum_p weight[p, i, i]   (deterministic 2-stage, fixed order)
// ---------------------------------------------------------------------------
__global__ void __launch_bounds__(256)
diagw_kernel(const float* __restrict__ w, float* __restrict__ dgw) {
    __shared__ float part[4][FD];
    const int i = threadIdx.x & 63;
    const int c = threadIdx.x >> 6;                 // 0..3, 500 rows each
    float s = 0.f;
    for (int p = c * 500; p < (c + 1) * 500; ++p)
        s += w[(size_t)p * FD * FD + (size_t)i * FD + i];
    part[c][i] = s;
    __syncthreads();
    if (threadIdx.x < FD)
        dgw[threadIdx.x] = part[0][threadIdx.x] + part[1][threadIdx.x] +
                           part[2][threadIdx.x] + part[3][threadIdx.x];
}

// ---------------------------------------------------------------------------
// z1[t][n][f] = x[n][f][t] / max(||x[n][:][t]||, eps), stored as f16 row-major
// one wave per (n, t)
// ---------------------------------------------------------------------------
__global__ void __launch_bounds__(256)
normalize_kernel(const float* __restrict__ x, _Float16* __restrict__ z1) {
    const int wid  = blockIdx.x * (blockDim.x >> 5) + (threadIdx.x >> 5);
    const int lane = threadIdx.x & 31;
    if (wid >= N_NODES * TT) return;
    const int n = wid / TT;
    const int t = wid % TT;

    const float* xr = x + (size_t)n * FD * TT + t;      // stride TT per f
    const float a = xr[(size_t)lane * TT];
    const float b = xr[(size_t)(lane + 32) * TT];
    float ss = a * a + b * b;
    ss += __shfl_xor(ss, 1,  32);
    ss += __shfl_xor(ss, 2,  32);
    ss += __shfl_xor(ss, 4,  32);
    ss += __shfl_xor(ss, 8,  32);
    ss += __shfl_xor(ss, 16, 32);
    const float scale = 1.0f / fmaxf(sqrtf(ss), 1e-12f);

    _Float16* zr = z1 + ((size_t)t * N_NODES + n) * FD;
    zr[lane]      = (_Float16)(a * scale);
    zr[lane + 32] = (_Float16)(b * scale);
}

// ---------------------------------------------------------------------------
// For each (t, n): ds[t][n] = softmax(Z1_t Z1_t^T)[n][n]
// One wave handles 16 rows (one M-tile), sweeps all 125 column tiles with
// v_wmma_f32_16x16x32_f16 (K=64 -> 2 WMMAs/tile). Rows are unit vectors so
// scores are in [-1,1]: fixed shift exp(s-1) replaces the running max, and
// the diagonal s_nn is computed directly from the A-fragment registers
// (sum of f16 squares), keeping the tile loop branch-free (EXEC all-ones).
// ---------------------------------------------------------------------------
__global__ void __launch_bounds__(128)
score_diag_kernel(const _Float16* __restrict__ z1, float* __restrict__ ds) {
    const int wave = blockIdx.x * (blockDim.x >> 5) + (threadIdx.x >> 5);
    if (wave >= TT * NTILES) return;
    const int lane    = threadIdx.x & 31;
    const int lane_lo = lane & 15;
    const int hi      = lane >> 4;
    const int t  = wave / NTILES;
    const int n0 = (wave % NTILES) << 4;

    const _Float16* zt = z1 + (size_t)t * N_NODES * FD;

    // A fragments for rows n0..n0+15 (A 16x32 layout: K = (e&7) + 8*hi + 16*(e>=8))
    const _Float16* arow = zt + (size_t)(n0 + lane_lo) * FD;
    Frag16 a0, a1;
    a0.h[0] = *(const v8h*)(arow +       8 * hi);
    a0.h[1] = *(const v8h*)(arow + 16 +  8 * hi);
    a1.h[0] = *(const v8h*)(arow + 32 +  8 * hi);
    a1.h[1] = *(const v8h*)(arow + 48 +  8 * hi);

    // Diagonal s_nn for row (n0 + lane_lo): the two lane-halves together hold
    // all 64 K-values of the row, so sum f16^2 locally + one xor-16 shuffle.
    float p = 0.f;
#pragma unroll
    for (int e = 0; e < 16; ++e) {
        const float q0 = (float)a0.v[e];
        const float q1 = (float)a1.v[e];
        p += q0 * q0 + q1 * q1;
    }
    p += __shfl_xor(p, 16, 32);           // now: s_nn for row n0+lane_lo, all lanes

    float rsum[8];
#pragma unroll
    for (int r = 0; r < 8; ++r) rsum[r] = 0.f;

#pragma unroll 1
    for (int m0 = 0; m0 < N_NODES; m0 += 16) {
        // B fragments for cols m0..m0+15 (B 32x16 layout: K = (e&7) + 8*hi + 16*(e>=8))
        const _Float16* brow = zt + (size_t)(m0 + lane_lo) * FD;
        Frag16 b0, b1;
        b0.h[0] = *(const v8h*)(brow +       8 * hi);
        b0.h[1] = *(const v8h*)(brow + 16 +  8 * hi);
        b1.h[0] = *(const v8h*)(brow + 32 +  8 * hi);
        b1.h[1] = *(const v8h*)(brow + 48 +  8 * hi);

        v8f c = {};
        c = __builtin_amdgcn_wmma_f32_16x16x32_f16(false, a0.v, false, b0.v,
                                                   (short)0, c, false, false);
        c = __builtin_amdgcn_wmma_f32_16x16x32_f16(false, a1.v, false, b1.v,
                                                   (short)0, c, false, false);

        // C layout: element r of v8f = row (r + 8*hi), col (m0 + lane_lo)
#pragma unroll
        for (int r = 0; r < 8; ++r)
            rsum[r] += __expf(c[r] - 1.0f);
    }

    // reduce row sums across the 16 lanes of each half
#pragma unroll
    for (int r = 0; r < 8; ++r) {
        float s = rsum[r];
        s += __shfl_xor(s, 1, 32);
        s += __shfl_xor(s, 2, 32);
        s += __shfl_xor(s, 4, 32);
        s += __shfl_xor(s, 8, 32);
        rsum[r] = s;
    }
    if (lane_lo == 0) {
#pragma unroll
        for (int r = 0; r < 8; ++r) {
            const int row = r + 8 * hi;                       // 0..15
            const float snn = __shfl(p, row, 32);             // from lane 'row'
            ds[(size_t)t * N_NODES + n0 + row] = __expf(snn - 1.0f) / rsum[r];
        }
    }
}

// ---------------------------------------------------------------------------
// diag_adj[n] = adj_norm[n,n] * sum_t ds[t][n]    (fixed-order, deterministic)
// ---------------------------------------------------------------------------
__global__ void __launch_bounds__(256)
diagadj_kernel(const float* __restrict__ adj, const float* __restrict__ ds,
               float* __restrict__ dadj) {
    const int n = blockIdx.x * blockDim.x + threadIdx.x;
    if (n >= N_NODES) return;
    float s = 0.f;
    for (int t = 0; t < TT; ++t)
        s += ds[(size_t)t * N_NODES + n];
    dadj[n] = adj[(size_t)n * N_NODES + n] * s;
}

// ---------------------------------------------------------------------------
// Per node n: xt = x[n]^T (T,F); q/k/v = xt@W*; a = softmax(q k^T / 8);
// xf = a@v; out[n][f][t] = xf[t][f] * diag_adj[n] * diag_w[f]
// One 256-thread workgroup per node; everything staged in LDS.
// ---------------------------------------------------------------------------
__global__ void __launch_bounds__(256)
cross_attn_kernel(const float* __restrict__ x,
                  const float* __restrict__ Wq, const float* __restrict__ Wk,
                  const float* __restrict__ Wv,
                  const float* __restrict__ dadj, const float* __restrict__ dgw,
                  float* __restrict__ out) {
    __shared__ float xt[TT][FD];          // 6 KB
    __shared__ float qs[TT][FD];          // 6 KB
    __shared__ float ks[TT][FD];          // 6 KB
    __shared__ float vs[TT][FD];          // 6 KB
    __shared__ float att[TT][TT + 1];     // ~2.4 KB

    const int n   = blockIdx.x;
    const int tid = threadIdx.x;
    const float* xn = x + (size_t)n * FD * TT;

    // load + transpose x[n] : (F,T) -> (T,F)
    for (int idx = tid; idx < FD * TT; idx += 256) {
        const int f = idx / TT, t = idx % TT;
        xt[t][f] = xn[idx];
    }
    __syncthreads();

    // projections: thread (f = tid&63, tq = tid>>6) handles 6 t-rows for col f
    {
        const int f  = tid & 63;
        const int tq = tid >> 6;          // 0..3
        float aq[6], ak[6], av[6];
#pragma unroll
        for (int j = 0; j < 6; ++j) { aq[j] = 0.f; ak[j] = 0.f; av[j] = 0.f; }
        for (int g = 0; g < FD; ++g) {
            const float wq = Wq[(size_t)g * FD + f];
            const float wk = Wk[(size_t)g * FD + f];
            const float wv = Wv[(size_t)g * FD + f];
#pragma unroll
            for (int j = 0; j < 6; ++j) {
                const float xv = xt[tq * 6 + j][g];
                aq[j] += xv * wq;
                ak[j] += xv * wk;
                av[j] += xv * wv;
            }
        }
#pragma unroll
        for (int j = 0; j < 6; ++j) {
            qs[tq * 6 + j][f] = aq[j];
            ks[tq * 6 + j][f] = ak[j];
            vs[tq * 6 + j][f] = av[j];
        }
    }
    __syncthreads();

    // scores (24x24), scaled by 1/sqrt(64)
    for (int idx = tid; idx < TT * TT; idx += 256) {
        const int t = idx / TT, s = idx % TT;
        float acc = 0.f;
        for (int g = 0; g < FD; ++g)
            acc += qs[t][g] * ks[s][g];
        att[t][s] = acc * 0.125f;
    }
    __syncthreads();

    // row softmax (24 rows)
    if (tid < TT) {
        float mx = att[tid][0];
        for (int s = 1; s < TT; ++s) mx = fmaxf(mx, att[tid][s]);
        float sum = 0.f;
        for (int s = 0; s < TT; ++s) {
            const float e = __expf(att[tid][s] - mx);
            att[tid][s] = e;
            sum += e;
        }
        const float inv = 1.0f / sum;
        for (int s = 0; s < TT; ++s) att[tid][s] *= inv;
    }
    __syncthreads();

    // xf = att @ v, fused output scaling; out is (N, F, T)
    {
        const int f  = tid & 63;
        const int tq = tid >> 6;
        const float sn = dadj[n];
        const float sf = dgw[f] * sn;
#pragma unroll
        for (int j = 0; j < 6; ++j) {
            const int t = tq * 6 + j;
            float acc = 0.f;
            for (int s = 0; s < TT; ++s)
                acc += att[t][s] * vs[s][f];
            out[(size_t)n * FD * TT + (size_t)f * TT + t] = acc * sf;
        }
    }
}

// ---------------------------------------------------------------------------
extern "C" void kernel_launch(void* const* d_in, const int* in_sizes, int n_in,
                              void* d_out, int out_size, void* d_ws, size_t ws_size,
                              hipStream_t stream) {
    (void)in_sizes; (void)n_in; (void)out_size; (void)ws_size;
    const float* x   = (const float*)d_in[0];
    const float* adj = (const float*)d_in[1];
    const float* w   = (const float*)d_in[2];
    const float* Wq  = (const float*)d_in[3];
    const float* Wk  = (const float*)d_in[4];
    const float* Wv  = (const float*)d_in[5];
    float* out = (float*)d_out;

    char* ws = (char*)d_ws;
    _Float16* z1 = (_Float16*)ws;                         // 24*2000*64*2 = 6,144,000 B
    float* ds    = (float*)(ws + 6144000);                // 24*2000*4    =   192,000 B
    float* dadj  = (float*)(ws + 6336000);                // 2000*4       =     8,000 B
    float* dgw   = (float*)(ws + 6344192);                // 64*4         =       256 B

    diagw_kernel    <<<1,    256, 0, stream>>>(w, dgw);
    normalize_kernel<<<6000, 256, 0, stream>>>(x, z1);    // 48000 waves, 8/block
    score_diag_kernel<<<750, 128, 0, stream>>>(z1, ds);   // 3000 waves, 4/block
    diagadj_kernel  <<<8,    256, 0, stream>>>(adj, ds, dadj);
    cross_attn_kernel<<<N_NODES, 256, 0, stream>>>(x, Wq, Wk, Wv, dadj, dgw, out);
}
